// DeployModel_50534585205513
// MI455X (gfx1250) — compile-verified, matrix operand order checked by reading, not run on hardware
//
#include <hip/hip_runtime.h>
#include <stdint.h>

#define NUM_CLASSES 80
#define PRE_K 1000
#define KEEP_K 100
#define NBINS 4096
#define CAND_CAP 4096
#define NIMG 32
#define NSEL 1024
#define IOU_THR 0.65f
#define SCORE_THR 0.25f
#define CLS_OFF 4096.0f
#define AREA_SCL 0.015625f   /* 1/64: keep f16 area in range */

typedef _Float16 v16h __attribute__((ext_vector_type(16)));
typedef float    v8f  __attribute__((ext_vector_type(8)));

/* ---- workspace byte offsets ---- */
#define WS_HIST   0u                                   /* u32[32*4096]  = 524288 B */
#define WS_THR    524288u                              /* u32[32]                  */
#define WS_CNT    524416u                              /* u32[32]                  */
#define WS_KEYS   524544u                              /* u64[32*4096]  = 1 MiB    */
#define WS_BOXO   1573120u                             /* f32[32*1024*4]           */
#define WS_AREA   2097408u                             /* f32[32*1024]             */
#define WS_BOXP   2228480u                             /* f32[32*1024*4]           */
#define WS_SCORE  2752768u                             /* f32[32*1024]             */
#define WS_LABEL  2883840u                             /* u32[32*1024]             */
#define WS_VALID  3014912u                             /* u32[32*1024]             */
#define WS_ADJ    3145984u                             /* u32[32*1024*32] = 4 MiB  */
/* total: 7340288 bytes */

__device__ __forceinline__ float sigmoidf(float x) {
    return 1.0f / (1.0f + __expf(-x));
}

/* ---------------- K0: zero hist/thr/cnt + adjacency ---------------- */
__global__ void k_zero(unsigned* __restrict__ ws) {
    unsigned stride = gridDim.x * blockDim.x;
    unsigned tid = blockIdx.x * blockDim.x + threadIdx.x;
    for (unsigned t = tid; t < 131136u; t += stride) ws[t] = 0u;        /* hist+thr+cnt */
    unsigned* adj = ws + (WS_ADJ / 4u);
    for (unsigned t = tid; t < 1048576u; t += stride) adj[t] = 0u;      /* adjacency */
}

/* ---------------- K1: score histogram (streaming pass 1) ---------------- */
__global__ void k_hist(const float* __restrict__ cls, const float* __restrict__ obj,
                       int H, int W, unsigned* __restrict__ hist) {
    __shared__ unsigned sh[NBINS];
    for (int t = threadIdx.x; t < NBINS; t += blockDim.x) sh[t] = 0u;
    __syncthreads();

    unsigned idx = blockIdx.x * blockDim.x + threadIdx.x;
    unsigned hw = (unsigned)(H * W);
    unsigned per_b = 240u * hw;
    unsigned total = per_b * 32u;
    unsigned b  = idx / per_b;          /* uniform per block (per_b % 256 == 0) */
    unsigned r  = idx - b * per_b;
    unsigned ch = r / hw;
    unsigned pix = r - ch * hw;
    unsigned p  = ch / 80u;

    if (idx + 8192u < total) __builtin_prefetch(cls + idx + 8192u, 0, 0);

    float cv = cls[idx];
    float ov = obj[(b * 3u + p) * hw + pix];
    float s  = sigmoidf(cv) * sigmoidf(ov);
    unsigned bin = __float_as_uint(s) >> 19;        /* monotone for s in (0,1) */
    atomicAdd(&sh[bin & (NBINS - 1u)], 1u);
    __syncthreads();

    for (int t = threadIdx.x; t < NBINS; t += blockDim.x) {
        unsigned c = sh[t];
        if (c) atomicAdd(&hist[b * NBINS + (unsigned)t], c);
    }
}

/* ---------------- K2: find rank-PRE_K threshold bin per image ---------------- */
__global__ void k_scan(const unsigned* __restrict__ hist, unsigned* __restrict__ thr) {
    int img = blockIdx.x;
    const unsigned* h = hist + img * NBINS;
    __shared__ unsigned part[256];
    __shared__ unsigned above[256];
    int t = threadIdx.x;

    unsigned sum = 0;
    for (int k = 0; k < 16; k++) sum += h[t * 16 + k];
    part[t] = sum;
    __syncthreads();
    if (t == 0) {
        unsigned acc = 0;
        for (int u = 255; u >= 0; u--) { above[u] = acc; acc += part[u]; }
    }
    __syncthreads();
    unsigned ab = above[t];
    if (ab < PRE_K && ab + part[t] >= PRE_K) {
        unsigned cum = ab;
        for (int b = t * 16 + 15; b >= t * 16; b--) {
            cum += h[b];
            if (cum >= PRE_K) { thr[img] = (unsigned)b; break; }
        }
    }
}

/* ---------------- K1b: compact candidates >= threshold (streaming pass 2) ---------------- */
__global__ void k_collect(const float* __restrict__ cls, const float* __restrict__ obj,
                          int H, int W, unsigned base,
                          const unsigned* __restrict__ thr,
                          unsigned* __restrict__ cnt, uint64_t* __restrict__ keys) {
    unsigned idx = blockIdx.x * blockDim.x + threadIdx.x;
    unsigned hw = (unsigned)(H * W);
    unsigned per_b = 240u * hw;
    unsigned total = per_b * 32u;
    unsigned b  = idx / per_b;
    unsigned r  = idx - b * per_b;
    unsigned ch = r / hw;
    unsigned pix = r - ch * hw;
    unsigned p  = ch / 80u;
    unsigned c  = ch - p * 80u;

    if (idx + 8192u < total) __builtin_prefetch(cls + idx + 8192u, 0, 0);

    float cv = cls[idx];
    float ov = obj[(b * 3u + p) * hw + pix];
    float s  = sigmoidf(cv) * sigmoidf(ov);
    unsigned bin = __float_as_uint(s) >> 19;
    if (bin >= thr[b]) {
        unsigned pos = atomicAdd(&cnt[b], 1u);
        if (pos < CAND_CAP) {
            unsigned anchor = base + (pix * 3u + p);
            unsigned cand = anchor * 80u + c;
            /* desc sort key: score bits high, (~idx) low -> stable lowest-index-first on ties */
            keys[b * CAND_CAP + pos] =
                ((uint64_t)__float_as_uint(s) << 32) | (uint64_t)(0xFFFFFFFFu - cand);
        }
    }
}

/* ---------------- K3: per-image bitonic sort + box decode ---------------- */
__global__ void __launch_bounds__(1024)
k_sortdecode(const uint64_t* __restrict__ keys, const unsigned* __restrict__ cnt,
             const float* __restrict__ priors, const float* __restrict__ strides,
             const float* __restrict__ bb0, const float* __restrict__ bb1,
             const float* __restrict__ bb2,
             float* __restrict__ boxO, float* __restrict__ areaA,
             float* __restrict__ boxP, float* __restrict__ scoreA,
             unsigned* __restrict__ labelA, unsigned* __restrict__ validA) {
    int img = blockIdx.x;
    __shared__ uint64_t sk[CAND_CAP];
    unsigned n = cnt[img]; if (n > CAND_CAP) n = CAND_CAP;
    for (unsigned t = threadIdx.x; t < CAND_CAP; t += blockDim.x)
        sk[t] = (t < n) ? keys[(unsigned)img * CAND_CAP + t] : 0ull;
    __syncthreads();

    /* bitonic sort, descending */
    for (unsigned k = 2; k <= CAND_CAP; k <<= 1) {
        for (unsigned j = k >> 1; j > 0; j >>= 1) {
            for (unsigned t = threadIdx.x; t < CAND_CAP; t += blockDim.x) {
                unsigned ixj = t ^ j;
                if (ixj > t) {
                    uint64_t a = sk[t], bv = sk[ixj];
                    bool up = ((t & k) == 0u);
                    if (up ? (a < bv) : (a > bv)) { sk[t] = bv; sk[ixj] = a; }
                }
            }
            __syncthreads();
        }
    }

    /* decode top NSEL entries (threadIdx.x : 0..1023) */
    int jslot = threadIdx.x;
    uint64_t key = sk[jslot];
    float s = __uint_as_float((unsigned)(key >> 32));
    unsigned cand = 0xFFFFFFFFu - (unsigned)(key & 0xFFFFFFFFu);
    unsigned anchor = cand / 80u;
    unsigned lab = cand - anchor * 80u;
    if (anchor >= 25200u) anchor = 25199u;
    if (lab > 79u) lab = 79u;

    unsigned base, Hh, Ww; const float* bb;
    if (anchor < 19200u)      { base = 0u;     Hh = 80u; Ww = 80u; bb = bb0; }
    else if (anchor < 24000u) { base = 19200u; Hh = 40u; Ww = 40u; bb = bb1; }
    else                      { base = 24000u; Hh = 20u; Ww = 20u; bb = bb2; }
    unsigned al = anchor - base;
    unsigned p = al % 3u, cell = al / 3u;
    unsigned wx = cell % Ww, hy = cell / Ww;
    unsigned hw = Hh * Ww;
    unsigned o = ((unsigned)img * 12u + p * 4u) * hw + hy * Ww + wx;

    float p0 = sigmoidf(bb[o]);
    float p1 = sigmoidf(bb[o + hw]);
    float p2 = sigmoidf(bb[o + 2u * hw]);
    float p3 = sigmoidf(bb[o + 3u * hw]);

    float pr0 = priors[anchor * 4u + 0u], pr1 = priors[anchor * 4u + 1u];
    float pr2 = priors[anchor * 4u + 2u], pr3 = priors[anchor * 4u + 3u];
    float st = strides[anchor];
    float cx = (pr0 + pr2) * 0.5f, cy = (pr1 + pr3) * 0.5f;
    float pw = pr2 - pr0, ph = pr3 - pr1;
    float xc = (p0 - 0.5f) * 2.0f * st + cx;
    float yc = (p1 - 0.5f) * 2.0f * st + cy;
    float wp = p2 * 2.0f; wp = wp * wp * pw;
    float hp = p3 * 2.0f; hp = hp * hp * ph;
    float x1 = xc - wp * 0.5f, y1 = yc - hp * 0.5f;
    float x2 = xc + wp * 0.5f, y2 = yc + hp * 0.5f;

    unsigned gi = (unsigned)img * NSEL + (unsigned)jslot;
    boxP[gi * 4u + 0u] = x1; boxP[gi * 4u + 1u] = y1;
    boxP[gi * 4u + 2u] = x2; boxP[gi * 4u + 3u] = y2;
    float off = (float)lab * CLS_OFF;
    boxO[gi * 4u + 0u] = x1 + off; boxO[gi * 4u + 1u] = y1 + off;
    boxO[gi * 4u + 2u] = x2 + off; boxO[gi * 4u + 3u] = y2 + off;
    areaA[gi] = (x2 - x1) * (y2 - y1);
    scoreA[gi] = s;
    labelA[gi] = lab;
    validA[gi] = (jslot < PRE_K && s > SCORE_THR) ? 1u : 0u;
}

/* ---------------- K4: IoU adjacency via WMMA (pairwise area-sum on matrix pipe),
                    tables staged into LDS with async global->LDS DMA ---------------- */
__global__ void __launch_bounds__(256)
k_adj(const float* __restrict__ boxO, const float* __restrict__ areaA,
      unsigned* __restrict__ adj) {
    int img = blockIdx.x;
    __shared__ float sB[NSEL * 4];     /* 16 KB offset boxes */
    __shared__ float sA[NSEL];         /*  4 KB areas        */

    /* async global->LDS staging: ASYNCcnt-tracked, no VGPR round trip */
    {
        const float* gB = boxO + (size_t)img * (NSEL * 4);
        const float* gA = areaA + (size_t)img * NSEL;
        unsigned t = threadIdx.x;
        unsigned ldsB = (unsigned)(uintptr_t)(&sB[0]);
        unsigned ldsA = (unsigned)(uintptr_t)(&sA[0]);
#pragma unroll
        for (int k = 0; k < 4; k++) {
            unsigned off = (unsigned)(k * 256) * 16u + t * 16u;   /* coalesced b128 */
            unsigned dst = ldsB + off;
            asm volatile("global_load_async_to_lds_b128 %0, %1, %2"
                         :: "v"(dst), "v"(off), "s"(gB) : "memory");
        }
        {
            unsigned off = t * 16u;
            unsigned dst = ldsA + off;
            asm volatile("global_load_async_to_lds_b128 %0, %1, %2"
                         :: "v"(dst), "v"(off), "s"(gA) : "memory");
        }
        asm volatile("s_wait_asynccnt 0x0" ::: "memory");
    }
    __syncthreads();

    int wave = threadIdx.x >> 5;
    int lane = threadIdx.x & 31;

    for (int tt = 0; tt < 16; tt++) {
        int tile = blockIdx.y * 128 + wave * 16 + tt;   /* 4096 tiles = 64x64 of 16x16 */
        int I0 = (tile >> 6) << 4;
        int J0 = (tile & 63) << 4;
        if (J0 + 15 < I0) continue;                     /* only need j > i */

        /* D[i][j] = area_i + area_j  via  A = [area_i, 1, 0...], B = [1; area_j; 0...] */
        v16h af = {}; v16h bf = {};
        if (lane < 16) {
            af[0] = (_Float16)(sA[I0 + lane] * AREA_SCL);
            af[1] = (_Float16)1.0f;
            bf[0] = (_Float16)1.0f;
            bf[1] = (_Float16)(sA[J0 + lane] * AREA_SCL);
        }
        v8f c = {};
        v8f d = __builtin_amdgcn_wmma_f32_16x16x32_f16(
            false, af, false, bf, (short)0, c, false, false);

        int nj = J0 + (lane & 15);
        float jx1 = sB[nj * 4 + 0], jy1 = sB[nj * 4 + 1];
        float jx2 = sB[nj * 4 + 2], jy2 = sB[nj * 4 + 3];
        int mb = (lane >= 16) ? 8 : 0;
        unsigned wword = (unsigned)J0 >> 5;
        unsigned shft = (J0 & 16) ? 16u : 0u;

#pragma unroll
        for (int r = 0; r < 8; r++) {
            int i = I0 + mb + r;
            float ix1 = sB[i * 4 + 0], iy1 = sB[i * 4 + 1];
            float ix2 = sB[i * 4 + 2], iy2 = sB[i * 4 + 3];
            float ww = fminf(ix2, jx2) - fmaxf(ix1, jx1); ww = fmaxf(ww, 0.0f);
            float hh = fminf(iy2, jy2) - fmaxf(iy1, jy1); hh = fmaxf(hh, 0.0f);
            float inter = ww * hh;
            float sum = d[r] * 64.0f;                   /* undo AREA_SCL */
            /* iou > thr  <=>  inter*(1+thr) > thr*(sum + eps) */
            bool pred = inter * (1.0f + IOU_THR) > IOU_THR * (sum + 1e-7f);
            unsigned m = __builtin_amdgcn_ballot_w32(pred);
            if (lane == 0) {
                unsigned lo = m & 0xFFFFu, hi = m >> 16;
                if (lo) atomicOr(&adj[((unsigned)img * NSEL + (unsigned)(I0 + r)) * 32u + wword], lo << shft);
                if (hi) atomicOr(&adj[((unsigned)img * NSEL + (unsigned)(I0 + 8 + r)) * 32u + wword], hi << shft);
            }
        }
    }
}

/* ---------------- K5: greedy bitmask NMS + outputs (one wave / image) ---------------- */
__global__ void __launch_bounds__(32)
k_nms(const unsigned* __restrict__ adj, const unsigned* __restrict__ validA,
      const float* __restrict__ boxP, const float* __restrict__ scoreA,
      const unsigned* __restrict__ labelA, float* __restrict__ out) {
    int img = blockIdx.x;
    int lane = threadIdx.x;                 /* 0..31, lane owns supp word `lane` */

    unsigned supp = 0u, vw = 0u;
    for (int k = 0; k < 32; k++) {
        unsigned i = (unsigned)lane * 32u + (unsigned)k;
        unsigned v = validA[(unsigned)img * NSEL + i];
        if (!v) supp |= (1u << k);
        else    vw   |= (1u << k);
    }

    for (int i = 0; i < PRE_K; i++) {
        unsigned wsupp = (unsigned)__builtin_amdgcn_readlane((int)supp, i >> 5);
        if (!((wsupp >> (i & 31)) & 1u)) {
            unsigned row = adj[((unsigned)img * NSEL + (unsigned)i) * 32u + (unsigned)lane];
            unsigned iw = (unsigned)i >> 5, ib = (unsigned)i & 31u;
            unsigned m;
            if ((unsigned)lane > iw)       m = 0xFFFFFFFFu;
            else if ((unsigned)lane == iw) m = (ib == 31u) ? 0u : (0xFFFFFFFFu << (ib + 1u));
            else                           m = 0u;
            supp |= row & m;
        }
    }

    __shared__ unsigned keepw[32];
    keepw[lane] = vw & ~supp;
    __syncthreads();

    if (lane == 0) {
        float* onum = out;
        float* ob = out + 32;
        float* os = out + 32 + NIMG * KEEP_K * 4;
        float* ol = out + 32 + NIMG * KEEP_K * 4 + NIMG * KEEP_K;
        int cntk = 0;
        for (int i = 0; i < PRE_K; i++) {
            if ((keepw[i >> 5] >> (i & 31)) & 1u) {
                if (cntk < KEEP_K) {
                    unsigned gi = (unsigned)img * NSEL + (unsigned)i;
                    int sL = img * KEEP_K + cntk;
                    ob[sL * 4 + 0] = boxP[gi * 4u + 0u];
                    ob[sL * 4 + 1] = boxP[gi * 4u + 1u];
                    ob[sL * 4 + 2] = boxP[gi * 4u + 2u];
                    ob[sL * 4 + 3] = boxP[gi * 4u + 3u];
                    os[sL] = scoreA[gi];
                    ol[sL] = (float)labelA[gi];
                }
                cntk++;
            }
        }
        int num = cntk < KEEP_K ? cntk : KEEP_K;
        onum[img] = (float)num;
        int slot = num;
        for (int i = 0; i < PRE_K && slot < KEEP_K; i++) {
            if (!((keepw[i >> 5] >> (i & 31)) & 1u)) {      /* zero-score fillers, ascending idx */
                unsigned gi = (unsigned)img * NSEL + (unsigned)i;
                int sL = img * KEEP_K + slot;
                ob[sL * 4 + 0] = boxP[gi * 4u + 0u];
                ob[sL * 4 + 1] = boxP[gi * 4u + 1u];
                ob[sL * 4 + 2] = boxP[gi * 4u + 2u];
                ob[sL * 4 + 3] = boxP[gi * 4u + 3u];
                os[sL] = 0.0f;
                ol[sL] = (float)labelA[gi];
                slot++;
            }
        }
    }
}

extern "C" void kernel_launch(void* const* d_in, const int* in_sizes, int n_in,
                              void* d_out, int out_size, void* d_ws, size_t ws_size,
                              hipStream_t stream) {
    (void)in_sizes; (void)n_in; (void)out_size; (void)ws_size;
    const float* cls0 = (const float*)d_in[0];
    const float* cls1 = (const float*)d_in[1];
    const float* cls2 = (const float*)d_in[2];
    const float* bb0  = (const float*)d_in[3];
    const float* bb1  = (const float*)d_in[4];
    const float* bb2  = (const float*)d_in[5];
    const float* obj0 = (const float*)d_in[6];
    const float* obj1 = (const float*)d_in[7];
    const float* obj2 = (const float*)d_in[8];
    const float* priors  = (const float*)d_in[9];
    const float* strides = (const float*)d_in[10];

    char* ws = (char*)d_ws;
    unsigned* hist   = (unsigned*)(ws + WS_HIST);
    unsigned* thr    = (unsigned*)(ws + WS_THR);
    unsigned* cnt    = (unsigned*)(ws + WS_CNT);
    uint64_t* keys   = (uint64_t*)(ws + WS_KEYS);
    float*    boxO   = (float*)(ws + WS_BOXO);
    float*    areaA  = (float*)(ws + WS_AREA);
    float*    boxP   = (float*)(ws + WS_BOXP);
    float*    scoreA = (float*)(ws + WS_SCORE);
    unsigned* labelA = (unsigned*)(ws + WS_LABEL);
    unsigned* validA = (unsigned*)(ws + WS_VALID);
    unsigned* adj    = (unsigned*)(ws + WS_ADJ);

    k_zero<<<2048, 256, 0, stream>>>((unsigned*)ws);

    k_hist<<<49152000 / 256, 256, 0, stream>>>(cls0, obj0, 80, 80, hist);
    k_hist<<<12288000 / 256, 256, 0, stream>>>(cls1, obj1, 40, 40, hist);
    k_hist<<< 3072000 / 256, 256, 0, stream>>>(cls2, obj2, 20, 20, hist);

    k_scan<<<NIMG, 256, 0, stream>>>(hist, thr);

    k_collect<<<49152000 / 256, 256, 0, stream>>>(cls0, obj0, 80, 80, 0u,     thr, cnt, keys);
    k_collect<<<12288000 / 256, 256, 0, stream>>>(cls1, obj1, 40, 40, 19200u, thr, cnt, keys);
    k_collect<<< 3072000 / 256, 256, 0, stream>>>(cls2, obj2, 20, 20, 24000u, thr, cnt, keys);

    k_sortdecode<<<NIMG, 1024, 0, stream>>>(keys, cnt, priors, strides, bb0, bb1, bb2,
                                            boxO, areaA, boxP, scoreA, labelA, validA);

    k_adj<<<dim3(NIMG, 32), 256, 0, stream>>>(boxO, areaA, adj);

    k_nms<<<NIMG, 32, 0, stream>>>(adj, validA, boxP, scoreA, labelA, (float*)d_out);
}
